// BitLinear_63204738728238
// MI455X (gfx1250) — 8-bit matrix operand order repaired and checked by reading, compiled, not run on hardware
//
#include <hip/hip_runtime.h>

#define EPSQ 1e-8f

typedef __attribute__((ext_vector_type(8))) int v8i;

__global__ __launch_bounds__(256) void bitlinear_wquant(
    const float* __restrict__ W, const float* __restrict__ alpha,
    signed char* __restrict__ Wq, float* __restrict__ wrescale)
{
    const int o = blockIdx.x;
    const int t = threadIdx.x;
    const float* row = W + (size_t)o * 2048;

    float v[8];
    float s = 0.f;
#pragma unroll
    for (int i = 0; i < 8; ++i) {
        v[i] = row[t * 8 + i];
        s += fabsf(v[i]);
    }

    __shared__ float red[256];
    __shared__ float s_scale;
    red[t] = s;
    __syncthreads();
#pragma unroll
    for (int off = 128; off > 0; off >>= 1) {
        if (t < off) red[t] += red[t + off];
        __syncthreads();
    }
    if (t == 0) {
        float sc = red[0] * (1.f / 2048.f) + EPSQ;
        s_scale = sc;
        wrescale[o] = alpha[o] * sc;
    }
    __syncthreads();
    const float inv = 1.f / s_scale;

    int p0 = 0, p1 = 0;
#pragma unroll
    for (int i = 0; i < 8; ++i) {
        float r = rintf(v[i] * inv);
        r = fminf(fmaxf(r, -1.f), 1.f);
        int q = (int)r & 0xff;
        if (i < 4) p0 |= q << (8 * i);
        else       p1 |= q << (8 * (i - 4));
    }
    int* dst = (int*)(Wq + (size_t)o * 2048 + t * 8);
    dst[0] = p0;
    dst[1] = p1;
}

__global__ __launch_bounds__(256) void bitlinear_aquant(
    const float* __restrict__ X, signed char* __restrict__ Aq,
    float* __restrict__ arescale)
{
    const int tok = blockIdx.x;
    const int t = threadIdx.x;
    const float* row = X + (size_t)tok * 2048;

    float v[8];
    float mx = 0.f;
#pragma unroll
    for (int i = 0; i < 8; ++i) {
        v[i] = row[t * 8 + i];
        mx = fmaxf(mx, fabsf(v[i]));
    }

    __shared__ float red[256];
    __shared__ float s_scale;
    red[t] = mx;
    __syncthreads();
#pragma unroll
    for (int off = 128; off > 0; off >>= 1) {
        if (t < off) red[t] = fmaxf(red[t], red[t + off]);
        __syncthreads();
    }
    if (t == 0) {
        float sc = red[0] + EPSQ;
        s_scale = sc;
        arescale[tok] = sc * (1.f / 127.f);
    }
    __syncthreads();
    const float k = 127.f / s_scale;

    int p0 = 0, p1 = 0;
#pragma unroll
    for (int i = 0; i < 8; ++i) {
        float r = v[i] * k;
        r = fminf(fmaxf(r, -128.f), 127.f);
        int q = (int)rintf(r) & 0xff;
        if (i < 4) p0 |= q << (8 * i);
        else       p1 |= q << (8 * (i - 4));
    }
    int* dst = (int*)(Aq + (size_t)tok * 2048 + t * 8);
    dst[0] = p0;
    dst[1] = p1;
}

__device__ __forceinline__ void gemm_kstep(
    const signed char* __restrict__ buf,
    const signed char* __restrict__ ap,
    int l16, int half, v8i* acc)
{
    int2 a01 = *(const int2*)(ap);
    int2 a23 = *(const int2*)(ap + 8);
    int2 a45 = *(const int2*)(ap + 32);
    int2 a67 = *(const int2*)(ap + 40);
    v8i a;
    a[0] = a01.x; a[1] = a01.y;
    a[2] = a23.x; a[3] = a23.y;
    a[4] = a45.x; a[5] = a45.y;
    a[6] = a67.x; a[7] = a67.y;

#pragma unroll
    for (int j = 0; j < 8; ++j) {
        const signed char* bp = buf + (j * 16 + l16) * 64 + half * 16;
        int4 b03 = *(const int4*)(bp);
        int4 b47 = *(const int4*)(bp + 32);
        v8i b;
        b[0] = b03.x; b[1] = b03.y; b[2] = b03.z; b[3] = b03.w;
        b[4] = b47.x; b[5] = b47.y; b[6] = b47.z; b[7] = b47.w;
        acc[j] = __builtin_amdgcn_wmma_i32_16x16x64_iu8(
             true, a,  true, b, acc[j],
             false,  false);
    }
}

__global__ __launch_bounds__(256) void bitlinear_gemm(
    const signed char* __restrict__ Aq, const signed char* __restrict__ Wq,
    const float* __restrict__ arescale, const float* __restrict__ wrescale,
    float* __restrict__ out)
{
    __shared__ alignas(16) signed char ldsB[2][128 * 64];

    const int lane  = threadIdx.x & 31;
    const int wave  = threadIdx.x >> 5;
    const int half  = lane >> 4;
    const int l16   = lane & 15;
    const int nBase = blockIdx.x * 128;
    const int mBase = blockIdx.y * 128 + wave * 16;

    v8i acc[8] = {};

    const signed char* arow = Aq + (size_t)(mBase + l16) * 2048;

    const int nStage   = threadIdx.x >> 1;
    const int segStage = threadIdx.x & 1;
    const signed char* bsrc = Wq + (size_t)(nBase + nStage) * 2048 + segStage * 32;
    const int stageOff = nStage * 64 + segStage * 32;

    int4 rA0 = ((const int4*)(bsrc))[0];
    int4 rB0 = ((const int4*)(bsrc))[1];
    int4 rA1 = ((const int4*)(bsrc + 64))[0];
    int4 rB1 = ((const int4*)(bsrc + 64))[1];
    ((int4*)(ldsB[0] + stageOff))[0] = rA0;
    ((int4*)(ldsB[0] + stageOff))[1] = rB0;
    __syncthreads();

    for (int itp = 0; itp < 16; ++itp) {
        const int it0 = itp * 2;

        ((int4*)(ldsB[1] + stageOff))[0] = rA1;
        ((int4*)(ldsB[1] + stageOff))[1] = rB1;
        if (it0 + 2 < 32) {
            rA0 = ((const int4*)(bsrc + (it0 + 2) * 64))[0];
            rB0 = ((const int4*)(bsrc + (it0 + 2) * 64))[1];
        }
        gemm_kstep(ldsB[0], arow + it0 * 64 + half * 16, l16, half, acc);
        __syncthreads();

        if (it0 + 2 < 32) {
            ((int4*)(ldsB[0] + stageOff))[0] = rA0;
            ((int4*)(ldsB[0] + stageOff))[1] = rB0;
        }
        if (it0 + 3 < 32) {
            rA1 = ((const int4*)(bsrc + (it0 + 3) * 64))[0];
            rB1 = ((const int4*)(bsrc + (it0 + 3) * 64))[1];
        }
        gemm_kstep(ldsB[1], arow + (it0 + 1) * 64 + half * 16, l16, half, acc);
        __syncthreads();
    }

    float am[8];
#pragma unroll
    for (int r = 0; r < 8; ++r)
        am[r] = arescale[mBase + 8 * half + r];

#pragma unroll
    for (int j = 0; j < 8; ++j) {
        const int n = nBase + j * 16 + l16;
        const float wr = wrescale[n];
#pragma unroll
        for (int r = 0; r < 8; ++r) {
            out[(size_t)(mBase + 8 * half + r) * 8192 + n] =
                (float)acc[j][r] * wr * am[r];
        }
    }
}

extern "C" void kernel_launch(void* const* d_in, const int* in_sizes, int n_in,
                              void* d_out, int out_size, void* d_ws, size_t ws_size,
                              hipStream_t stream)
{
    (void)in_sizes; (void)n_in; (void)out_size; (void)ws_size;

    const float* x     = (const float*)d_in[0];
    const float* W     = (const float*)d_in[1];
    const float* alpha = (const float*)d_in[2];
    float* out = (float*)d_out;

    char* ws = (char*)d_ws;
    signed char* Wq       = (signed char*)ws;
    signed char* Aq       = (signed char*)(ws + (size_t)16 * 1024 * 1024);
    float*       wrescale = (float*)(ws + (size_t)32 * 1024 * 1024);
    float*       arescale = (float*)(ws + (size_t)32 * 1024 * 1024 + 32768);

    bitlinear_wquant<<<8192, 256, 0, stream>>>(W, alpha, Wq, wrescale);
    bitlinear_aquant<<<8192, 256, 0, stream>>>(x, Aq, arescale);

    dim3 grid(8192 / 128, 8192 / 128);
    bitlinear_gemm<<<grid, 256, 0, stream>>>(Aq, Wq, arescale, wrescale, out);
}
